// PerformerWrapper_60679297958060
// MI455X (gfx1250) — matrix operand order repaired and checked by reading, not compiled, not run on hardware
//
#include <hip/hip_runtime.h>
#include <hip/hip_bf16.h>
#include <stdint.h>

// Performer (FAVOR+) attention, MI455X / gfx1250.
// bf16 WMMA pipeline + async global->LDS (ASYNCcnt) double-buffered GEMMs.
// B=8 S=4096 DIM=768 H=12 DH=64 M=256.

#define BATCH 8
#define SEQ   4096
#define DIMM  768
#define HEADS 12
#define DH    64
#define MF    256
#define ROWS  (BATCH*SEQ)
#define BH    (BATCH*HEADS)

#define DATA_NORM 0.35355339059327373f  // 64^-0.25
#define RATIO     0.0625f               // 256^-0.5
#define FEPS      1e-4f

typedef __attribute__((ext_vector_type(16))) __bf16 v16bf;
typedef __attribute__((ext_vector_type(8)))  float  v8f;

union FragAB { v16bf v; uint32_t u[8]; };
union FragC  { v8f v; float f[8]; };

__device__ __forceinline__ uint16_t f2bf(float f) {
  union { float f; uint32_t u; } x; x.f = f;
  uint32_t r = x.u + 0x7fffu + ((x.u >> 16) & 1u);   // RNE
  return (uint16_t)(r >> 16);
}
__device__ __forceinline__ float bf2f(uint16_t h) {
  union { uint32_t u; float f; } x; x.u = ((uint32_t)h) << 16;
  return x.f;
}
__device__ __forceinline__ uint32_t pack2(float a, float b) {
  return (uint32_t)f2bf(a) | ((uint32_t)f2bf(b) << 16);
}

__device__ __forceinline__ v8f wmma_bf16(v16bf a, v16bf b, v8f c) {
  return __builtin_amdgcn_wmma_f32_16x16x32_bf16(false, a, false, b, (short)0, c,
                                                 false, false);
}

// ---------------- async global -> LDS (CDNA5 GLOBAL_LOAD_ASYNC_TO_LDS_B128) ----------------
__device__ __forceinline__ uint32_t lds_off(const void* p) {
  return (uint32_t)(uintptr_t)(__attribute__((address_space(3))) const void*)p;
}
__device__ __forceinline__ void async_b128(uint32_t ldsaddr, const void* gaddr) {
  asm volatile("global_load_async_to_lds_b128 %0, %1, off"
               :: "v"(ldsaddr), "v"((unsigned long long)(uintptr_t)gaddr)
               : "memory");
}
__device__ __forceinline__ void async_wait() {
  asm volatile("s_wait_asynccnt 0x0" ::: "memory");
}

// ---- fragment loaders (wave32, 16x16x32 bf16 layouts per CDNA5 ISA 7.12.2) ----
// A fragment: tile stored [i][k] row-major (element stride `stride`, even).
__device__ __forceinline__ v16bf load_fragA(const uint16_t* t, int stride, int i0, int k0) {
  const int lane = threadIdx.x & 31;
  const uint32_t* row = (const uint32_t*)(t + (size_t)(i0 + (lane & 15)) * stride + k0);
  const int half = lane >> 4;
  FragAB f;
#pragma unroll
  for (int j = 0; j < 8; ++j) {
    int kk = (j >> 2) * 16 + half * 8 + (j & 3) * 2;
    f.u[j] = row[kk >> 1];
  }
  return f.v;
}
// B fragment: tile stored [n][k]; same K order as A (lanes 0-15: K 0-7,16-23; lanes 16-31: K 8-15,24-31).
__device__ __forceinline__ v16bf load_fragB(const uint16_t* t, int stride, int n0, int k0) {
  const int lane = threadIdx.x & 31;
  const uint32_t* row = (const uint32_t*)(t + (size_t)(n0 + (lane & 15)) * stride + k0);
  const int half = lane >> 4;
  FragAB f;
#pragma unroll
  for (int j = 0; j < 8; ++j) f.u[j] = row[(j >> 2) * 8 + half * 4 + (j & 3)];
  return f.v;
}

__device__ __forceinline__ void atomicMaxF(float* addr, float val) {
  if (val >= 0.f) atomicMax((int*)addr, __float_as_int(val));
  else            atomicMin((unsigned int*)addr, (unsigned int)__float_as_int(val));
}

// ================= prep: x -> bf16, W -> bf16 transposed [n][k] =================
__global__ __launch_bounds__(256)
void cvt_bf16_4(const float4* __restrict__ src, uint32_t* __restrict__ dst2, int n4) {
  int i = blockIdx.x * 256 + threadIdx.x;
  if (i >= n4) return;
  float4 v = src[i];
  dst2[i * 2]     = pack2(v.x, v.y);
  dst2[i * 2 + 1] = pack2(v.z, v.w);
}

__global__ __launch_bounds__(256)
void wt_prep(const float* __restrict__ W, uint16_t* __restrict__ WT) {
  // WT[n][k] = bf16(W[k][n]); 768x768, 32x32 tiles
  __shared__ float t[32][33];
  const int lx = threadIdx.x & 31, ly = threadIdx.x >> 5;
  const int kb = blockIdx.y * 32, nb = blockIdx.x * 32;
#pragma unroll
  for (int i = 0; i < 4; ++i)
    t[ly + i * 8][lx] = W[(size_t)(kb + ly + i * 8) * DIMM + nb + lx];
  __syncthreads();
#pragma unroll
  for (int i = 0; i < 4; ++i)
    WT[(size_t)(nb + ly + i * 8) * DIMM + kb + lx] = f2bf(t[lx][ly + i * 8]);
}

// ======== GEMM: Y[ROWS x 768] = X(bf16) @ W, WT[n][k] bf16; async double-buffered ========
template<bool OUT_F32>
__global__ __launch_bounds__(256)
void gemm768b(const uint16_t* __restrict__ X, const uint16_t* __restrict__ WT,
              void* __restrict__ Yv) {
  constexpr int LSA = 40;                      // padded stride: 80 B (16B-aligned)
  __shared__ __align__(16) uint16_t As[2][128 * LSA];
  __shared__ __align__(16) uint16_t Bs[2][128 * LSA];
  const int tid = threadIdx.x;
  const int wave = tid >> 5;
  const int m0 = blockIdx.y * 128;
  const int n0 = blockIdx.x * 128;
  const int wm = (wave >> 2) * 64;
  const int wn = (wave & 3) * 32;
  FragC acc[4][2];
#pragma unroll
  for (int i = 0; i < 4; ++i)
#pragma unroll
    for (int j = 0; j < 2; ++j)
#pragma unroll
      for (int q = 0; q < 8; ++q) acc[i][j].f[q] = 0.f;

  auto issue = [&](int kt, int buf) {
#pragma unroll
    for (int it = 0; it < 2; ++it) {
      int c = tid + it * 256;                  // 512 chunks of 16 B per tile
      int r = c >> 2, seg = (c & 3) * 8;
      async_b128(lds_off(&As[buf][r * LSA + seg]),
                 X + (size_t)(m0 + r) * DIMM + kt + seg);
      async_b128(lds_off(&Bs[buf][r * LSA + seg]),
                 WT + (size_t)(n0 + r) * DIMM + kt + seg);
    }
  };

  issue(0, 0);
  async_wait();
  __syncthreads();
  int buf = 0;
  for (int kt = 0; kt < DIMM; kt += 32, buf ^= 1) {
    if (kt + 32 < DIMM) issue(kt + 32, buf ^ 1);
    v16bf afr[4], bfr[2];
#pragma unroll
    for (int i = 0; i < 4; ++i) afr[i] = load_fragA(As[buf], LSA, wm + i * 16, 0);
#pragma unroll
    for (int j = 0; j < 2; ++j) bfr[j] = load_fragB(Bs[buf], LSA, wn + j * 16, 0);
#pragma unroll
    for (int i = 0; i < 4; ++i)
#pragma unroll
      for (int j = 0; j < 2; ++j)
        acc[i][j].v = wmma_bf16(afr[i], bfr[j], acc[i][j].v);
    async_wait();
    __syncthreads();
  }
  const int lane = tid & 31;
  const int cn = lane & 15;
  const int cm = (lane >> 4) * 8;
#pragma unroll
  for (int i = 0; i < 4; ++i)
#pragma unroll
    for (int j = 0; j < 2; ++j) {
      int rb = m0 + wm + i * 16 + cm;
      int cb = n0 + wn + j * 16 + cn;
#pragma unroll
      for (int q = 0; q < 8; ++q) {
        float val = acc[i][j].f[q];
        if (OUT_F32) ((float*)Yv)[(size_t)(rb + q) * DIMM + cb] = val;
        else ((uint16_t*)Yv)[(size_t)(rb + q) * DIMM + cb] = f2bf(val);
      }
    }
}

// ================= diag = 0.5 * dn^2 * sum_d data^2, per (b,h,s) =================
__global__ __launch_bounds__(256)
void row_sqnorm(const uint16_t* __restrict__ Q, float* __restrict__ diag) {
  int idx = blockIdx.x * 256 + threadIdx.x;   // row*HEADS + h
  if (idx >= ROWS * HEADS) return;
  int row = idx / HEADS, h = idx % HEADS;
  const uint32_t* p = (const uint32_t*)(Q + (size_t)row * DIMM + h * DH);
  float s = 0.f;
#pragma unroll
  for (int i = 0; i < DH / 2; ++i) {
    uint32_t w = p[i];
    float a = bf2f((uint16_t)w), b = bf2f((uint16_t)(w >> 16));
    s += a * a + b * b;
  }
  int b_ = row / SEQ, s_ = row % SEQ;
  diag[((size_t)b_ * HEADS + h) * SEQ + s_] = 0.5f * DATA_NORM * DATA_NORM * s;
}

// ================= projb = bf16(dn * proj), [256 x 64] =================
__global__ void proj_prep(const float* __restrict__ proj, uint16_t* __restrict__ projb) {
  int i = blockIdx.x * 256 + threadIdx.x;
  if (i < MF * DH) projb[i] = f2bf(proj[i] * DATA_NORM);
}

__global__ void init_kmax(float* kmax) {
  int i = threadIdx.x;
  if (i < BH) kmax[i] = -__builtin_inff();
}

// ---- async tile loaders into padded LDS (strides 72 elems = 144 B, 16B-aligned) ----
__device__ __forceinline__ void load_tile_128x64_async(const uint16_t* __restrict__ src,
                                                       int r0, int c0, uint16_t* dst) {
  const int tid = threadIdx.x;
#pragma unroll
  for (int it = 0; it < 4; ++it) {
    int c = tid + it * 256;                   // 1024 chunks of 16 B
    int r = c >> 3, seg = (c & 7) * 8;
    async_b128(lds_off(dst + r * 72 + seg),
               src + (size_t)(r0 + r) * DIMM + c0 + seg);
  }
}
__device__ __forceinline__ void load_projb_async(const uint16_t* __restrict__ pb,
                                                 uint16_t* dst) {
  const int tid = threadIdx.x;
#pragma unroll
  for (int it = 0; it < 8; ++it) {
    int c = tid + it * 256;                   // 2048 chunks of 16 B
    int r = c >> 3, seg = (c & 7) * 8;
    async_b128(lds_off(dst + r * 72 + seg), pb + (size_t)r * DH + seg);
  }
}

// dd frags: wave w computes rows [w*16, w*16+16) x all 256 feats, K = 64.
__device__ __forceinline__ void compute_dd(const uint16_t* kt, const uint16_t* pj,
                                           FragC dd[16]) {
  const int wave = threadIdx.x >> 5;
#pragma unroll
  for (int nf = 0; nf < 16; ++nf)
#pragma unroll
    for (int q = 0; q < 8; ++q) dd[nf].f[q] = 0.f;
#pragma unroll
  for (int kc = 0; kc < DH; kc += 32) {
    v16bf a = load_fragA(kt, 72, wave * 16, kc);
#pragma unroll
    for (int nf = 0; nf < 16; ++nf) {
      v16bf b = load_fragB(pj, 72, nf * 16, kc);
      dd[nf].v = wmma_bf16(a, b, dd[nf].v);
    }
  }
}

// ================= pass 1 over K: global per-head max of data_dash =================
__global__ __launch_bounds__(256)
void kmax_pass(const uint16_t* __restrict__ kb, const uint16_t* __restrict__ projb,
               float* __restrict__ kmax) {
  __shared__ __align__(16) uint16_t pj[MF * 72];
  __shared__ __align__(16) uint16_t kt[128 * 72];
  __shared__ float wmax[8];
  const int bh = blockIdx.y;
  const int b = bh / HEADS, h = bh % HEADS;
  const int r0 = b * SEQ + blockIdx.x * 128;
  load_projb_async(projb, pj);
  load_tile_128x64_async(kb, r0, h * DH, kt);
  async_wait();
  __syncthreads();
  FragC dd[16];
  compute_dd(kt, pj, dd);
  float m = -__builtin_inff();
#pragma unroll
  for (int nf = 0; nf < 16; ++nf)
#pragma unroll
    for (int q = 0; q < 8; ++q) m = fmaxf(m, dd[nf].f[q]);
#pragma unroll
  for (int off = 16; off >= 1; off >>= 1) m = fmaxf(m, __shfl_xor(m, off, 32));
  if ((threadIdx.x & 31) == 0) wmax[threadIdx.x >> 5] = m;
  __syncthreads();
  if (threadIdx.x == 0) {
    float mm = wmax[0];
    for (int i = 1; i < 8; ++i) mm = fmaxf(mm, wmax[i]);
    atomicMaxF(&kmax[bh], mm);
  }
}

// ====== pass 2 over K: kp = ratio*(exp(dd-diag-stab)+eps); ctx += kp^T v; ksum ======
__global__ __launch_bounds__(256)
void kp_ctx_pass(const uint16_t* __restrict__ kb, const uint16_t* __restrict__ vb,
                 const uint16_t* __restrict__ projb,
                 const float* __restrict__ diagk, const float* __restrict__ kmax,
                 float* __restrict__ ctx, float* __restrict__ ksum) {
  __shared__ __align__(16) uint16_t pj[MF * 72];     // 36864 B
  __shared__ __align__(16) uint16_t kt[128 * 72];    // 18432 B
  __shared__ __align__(16) uint16_t kpT[MF * 136];   // 69632 B  [feat][s]
  __shared__ __align__(16) uint16_t vT[DH * 136];    // 17408 B  [d][s]
  __shared__ float dg[128];
  const int bh = blockIdx.y;
  const int b = bh / HEADS, h = bh % HEADS;
  const int c0 = h * DH;
  const int tid = threadIdx.x, wave = tid >> 5, lane = tid & 31;
  const float stab = kmax[bh];
  load_projb_async(projb, pj);
  const int fb = (wave >> 1) * 64;   // 4 feat blocks of 64
  const int db = (wave & 1) * 32;    // 2 d blocks of 32
  FragC cacc[4][2];
#pragma unroll
  for (int i = 0; i < 4; ++i)
#pragma unroll
    for (int j = 0; j < 2; ++j)
#pragma unroll
      for (int q = 0; q < 8; ++q) cacc[i][j].f[q] = 0.f;
  float ksacc = 0.f;

  for (int sc = 0; sc < 8; ++sc) {
    const int soff = blockIdx.x * 1024 + sc * 128;
    const int r0 = b * SEQ + soff;
    __syncthreads();                                   // LDS reuse barrier
    load_tile_128x64_async(kb, r0, c0, kt);
    if (tid < 128) dg[tid] = diagk[(size_t)bh * SEQ + soff + tid];
#pragma unroll
    for (int it = 0; it < 16; ++it) {                  // vT[d][s] transpose
      int e = (tid + it * 256) * 2;                    // 8192 elems
      int r = e >> 6, c = e & 63;
      uint32_t w = *(const uint32_t*)(vb + (size_t)(r0 + r) * DIMM + c0 + c);
      vT[c * 136 + r]       = (uint16_t)w;
      vT[(c + 1) * 136 + r] = (uint16_t)(w >> 16);
    }
    async_wait();
    __syncthreads();
    FragC dd[16];
    compute_dd(kt, pj, dd);
    const int rbase = wave * 16 + ((lane >> 4) << 3);
#pragma unroll
    for (int nf = 0; nf < 16; ++nf) {
      int feat = nf * 16 + (lane & 15);
#pragma unroll
      for (int q = 0; q < 8; ++q) {
        int rl = rbase + q;
        float kv = RATIO * (__expf(dd[nf].f[q] - dg[rl] - stab) + FEPS);
        kpT[feat * 136 + rl] = f2bf(kv);
      }
    }
    __syncthreads();
    {   // ksum partial: thread handles feat = tid
      const uint16_t* rowp = kpT + tid * 136;
      float s = 0.f;
#pragma unroll 8
      for (int sI = 0; sI < 128; ++sI) s += bf2f(rowp[sI]);
      ksacc += s;
    }
    // ctx partial: [256 feat x 64 d] += kpT[feat][s] * vT[d][s], K = 128
#pragma unroll
    for (int kc = 0; kc < 128; kc += 32) {
      v16bf bfr[2];
#pragma unroll
      for (int j = 0; j < 2; ++j) bfr[j] = load_fragB(vT, 136, db + j * 16, kc);
#pragma unroll
      for (int i = 0; i < 4; ++i) {
        v16bf a = load_fragA(kpT, 136, fb + i * 16, kc);
#pragma unroll
        for (int j = 0; j < 2; ++j)
          cacc[i][j].v = wmma_bf16(a, bfr[j], cacc[i][j].v);
      }
    }
  }
  atomicAdd(&ksum[bh * MF + tid], ksacc);
  float* ctxh = ctx + (size_t)bh * MF * DH;
  const int cn = lane & 15, cm = (lane >> 4) * 8;
#pragma unroll
  for (int i = 0; i < 4; ++i)
#pragma unroll
    for (int j = 0; j < 2; ++j) {
      int feat0 = fb + i * 16 + cm;
      int d0 = db + j * 16 + cn;
#pragma unroll
      for (int q = 0; q < 8; ++q)
        atomicAdd(&ctxh[(feat0 + q) * DH + d0], cacc[i][j].f[q]);
    }
}

// ====== qp features + out = (qp @ ctx) * d_inv, store bf16 attn ======
__global__ __launch_bounds__(256)
void out_pass(const uint16_t* __restrict__ qb, const uint16_t* __restrict__ projb,
              const float* __restrict__ diagq, const float* __restrict__ ctx,
              const float* __restrict__ ksum, uint16_t* __restrict__ attn) {
  __shared__ __align__(16) uint16_t pj[MF * 72];      // 36864
  __shared__ __align__(16) uint16_t qt[128 * 72];     // 18432
  __shared__ __align__(16) uint16_t qp[128 * 264];    // 67584  [s][feat]
  __shared__ __align__(16) uint16_t ctxT[DH * 264];   // 33792  [d][feat]
  __shared__ float ks[MF];
  __shared__ float dg[128];
  __shared__ float dinv[128];
  const int bh = blockIdx.y;
  const int b = bh / HEADS, h = bh % HEADS;
  const int c0 = h * DH;
  const int tid = threadIdx.x, wave = tid >> 5, lane = tid & 31;
  const int soff = blockIdx.x * 128;
  const int r0 = b * SEQ + soff;
  load_projb_async(projb, pj);
  load_tile_128x64_async(qb, r0, c0, qt);
  if (tid < 128) dg[tid] = diagq[(size_t)bh * SEQ + soff + tid];
  ks[tid] = ksum[bh * MF + tid];
  {
    const float* ch = ctx + (size_t)bh * MF * DH;
#pragma unroll 8
    for (int it = 0; it < 64; ++it) {
      int e = tid + it * 256;           // 16384: ch[feat*64 + d]
      int f_ = e >> 6, d_ = e & 63;
      ctxT[d_ * 264 + f_] = f2bf(ch[e]);
    }
  }
  async_wait();
  __syncthreads();
  FragC dd[16];
  compute_dd(qt, pj, dd);
  const int rbase = wave * 16 + ((lane >> 4) << 3);
  float rmax[8], rsum[8];
#pragma unroll
  for (int q = 0; q < 8; ++q) { rmax[q] = -__builtin_inff(); rsum[q] = 0.f; }
#pragma unroll
  for (int nf = 0; nf < 16; ++nf)
#pragma unroll
    for (int q = 0; q < 8; ++q) rmax[q] = fmaxf(rmax[q], dd[nf].f[q]);
#pragma unroll
  for (int q = 0; q < 8; ++q)
#pragma unroll
    for (int off = 1; off <= 8; off <<= 1)
      rmax[q] = fmaxf(rmax[q], __shfl_xor(rmax[q], off, 32));
#pragma unroll
  for (int nf = 0; nf < 16; ++nf) {
    int feat = nf * 16 + (lane & 15);
    float ksv = ks[feat];
#pragma unroll
    for (int q = 0; q < 8; ++q) {
      int rl = rbase + q;
      float v = RATIO * (__expf(dd[nf].f[q] - dg[rl] - rmax[q]) + FEPS);
      rsum[q] += v * ksv;
      qp[rl * 264 + feat] = f2bf(v);
    }
  }
#pragma unroll
  for (int q = 0; q < 8; ++q) {
#pragma unroll
    for (int off = 1; off <= 8; off <<= 1) rsum[q] += __shfl_xor(rsum[q], off, 32);
    if ((lane & 15) == 0) dinv[rbase + q] = 1.f / rsum[q];
  }
  __syncthreads();
  // GEMM3: wave w -> rows [w*16, +16), 64 cols, K = 256
  FragC oacc[4];
#pragma unroll
  for (int j = 0; j < 4; ++j)
#pragma unroll
    for (int q = 0; q < 8; ++q) oacc[j].f[q] = 0.f;
#pragma unroll
  for (int kc = 0; kc < MF; kc += 32) {
    v16bf a = load_fragA(qp, 264, wave * 16, kc);
#pragma unroll
    for (int j = 0; j < 4; ++j) {
      v16bf bf_ = load_fragB(ctxT, 264, j * 16, kc);
      oacc[j].v = wmma_bf16(a, bf_, oacc[j].v);
    }
  }
  const int cn = lane & 15, cm = (lane >> 4) * 8;
#pragma unroll
  for (int j = 0; j < 4; ++j) {
    int d0 = j * 16 + cn;
#pragma unroll
    for (int q = 0; q < 8; ++q) {
      int rl = wave * 16 + cm + q;
      float val = oacc[j].f[q] * dinv[rl];
      attn[(size_t)(r0 + rl) * DIMM + c0 + d0] = f2bf(val);
    }
  }
}

// ======================= host-side launch =======================
// Workspace layout (bytes); requires ws_size >= ~266 MB.
static constexpr size_t OFF_QB    = 0;                       // bf16 [ROWS][768]
static constexpr size_t OFF_KB    = 50331648;
static constexpr size_t OFF_VB    = 100663296;
static constexpr size_t OFF_ATTN  = 150994944;
static constexpr size_t OFF_CTX   = 201326592;               // f32 [BH][256][64]
static constexpr size_t CTX_BYTES = (size_t)BH * MF * DH * 4;
static constexpr size_t OFF_KSUM  = OFF_CTX + CTX_BYTES;     // f32 [BH][256]
static constexpr size_t KSUM_BYTES= (size_t)BH * MF * 4;
static constexpr size_t OFF_KMAX  = OFF_KSUM + KSUM_BYTES;   // f32 [BH]
static constexpr size_t OFF_DIAGQ = OFF_KMAX + 512;          // f32 [BH][SEQ]
static constexpr size_t OFF_DIAGK = OFF_DIAGQ + (size_t)BH * SEQ * 4;
static constexpr size_t OFF_PROJB = OFF_DIAGK + (size_t)BH * SEQ * 4;  // bf16 [256][64]
static constexpr size_t OFF_XB    = OFF_PROJB + 65536;       // bf16 [ROWS][768]
static constexpr size_t OFF_WTQ   = OFF_XB + (size_t)ROWS * DIMM * 2;  // bf16 [768][768] T
static constexpr size_t WT_BYTES  = (size_t)DIMM * DIMM * 2;
static constexpr size_t OFF_WTK   = OFF_WTQ + WT_BYTES;
static constexpr size_t OFF_WTV   = OFF_WTK + WT_BYTES;
static constexpr size_t OFF_WTO   = OFF_WTV + WT_BYTES;

extern "C" void kernel_launch(void* const* d_in, const int* in_sizes, int n_in,
                              void* d_out, int out_size, void* d_ws, size_t ws_size,
                              hipStream_t stream) {
  (void)in_sizes; (void)n_in; (void)out_size; (void)ws_size;
  const float* x    = (const float*)d_in[0];
  const float* Wq   = (const float*)d_in[1];
  const float* Wk   = (const float*)d_in[2];
  const float* Wv   = (const float*)d_in[3];
  const float* Wo   = (const float*)d_in[4];
  const float* proj = (const float*)d_in[5];
  float* out = (float*)d_out;

  char* ws = (char*)d_ws;
  uint16_t* qb    = (uint16_t*)(ws + OFF_QB);
  uint16_t* kb    = (uint16_t*)(ws + OFF_KB);
  uint16_t* vb    = (uint16_t*)(ws + OFF_VB);
  uint16_t* attn  = (uint16_t*)(ws + OFF_ATTN);
  float*    ctx   = (float*)(ws + OFF_CTX);
  float*    ksum  = (float*)(ws + OFF_KSUM);
  float*    kmax  = (float*)(ws + OFF_KMAX);
  float*    diagq = (float*)(ws + OFF_DIAGQ);
  float*    diagk = (float*)(ws + OFF_DIAGK);
  uint16_t* projb = (uint16_t*)(ws + OFF_PROJB);
  uint16_t* xb    = (uint16_t*)(ws + OFF_XB);
  uint16_t* wtq   = (uint16_t*)(ws + OFF_WTQ);
  uint16_t* wtk   = (uint16_t*)(ws + OFF_WTK);
  uint16_t* wtv   = (uint16_t*)(ws + OFF_WTV);
  uint16_t* wto   = (uint16_t*)(ws + OFF_WTO);

  hipMemsetAsync(ctx, 0, CTX_BYTES + KSUM_BYTES, stream);   // ctx + ksum zeroed
  init_kmax<<<1, 128, 0, stream>>>(kmax);
  proj_prep<<<(MF * DH + 255) / 256, 256, 0, stream>>>(proj, projb);

  // prep: x -> bf16, weights -> bf16 transposed
  int n4 = ROWS * DIMM / 4;
  cvt_bf16_4<<<(n4 + 255) / 256, 256, 0, stream>>>((const float4*)x, (uint32_t*)xb, n4);
  dim3 gt(DIMM / 32, DIMM / 32);   // (24, 24)
  wt_prep<<<gt, 256, 0, stream>>>(Wq, wtq);
  wt_prep<<<gt, 256, 0, stream>>>(Wk, wtk);
  wt_prep<<<gt, 256, 0, stream>>>(Wv, wtv);
  wt_prep<<<gt, 256, 0, stream>>>(Wo, wto);

  dim3 g1(DIMM / 128, ROWS / 128);   // (6, 256)
  gemm768b<false><<<g1, 256, 0, stream>>>(xb, wtq, qb);
  gemm768b<false><<<g1, 256, 0, stream>>>(xb, wtk, kb);
  gemm768b<false><<<g1, 256, 0, stream>>>(xb, wtv, vb);

  int nrh = (ROWS * HEADS + 255) / 256;
  row_sqnorm<<<nrh, 256, 0, stream>>>(qb, diagq);
  row_sqnorm<<<nrh, 256, 0, stream>>>(kb, diagk);

  kmax_pass<<<dim3(SEQ / 128, BH), 256, 0, stream>>>(kb, projb, kmax);
  kp_ctx_pass<<<dim3(SEQ / 1024, BH), 256, 0, stream>>>(kb, vb, projb, diagk, kmax,
                                                        ctx, ksum);
  out_pass<<<dim3(SEQ / 128, BH), 256, 0, stream>>>(qb, projb, diagq, ctx, ksum, attn);

  gemm768b<true><<<g1, 256, 0, stream>>>(attn, wto, out);
}
